// MHA_RoPE_63213328662926
// MI455X (gfx1250) — matrix operand order repaired and checked by reading, not compiled, not run on hardware
//
#include <hip/hip_runtime.h>
#include <hip/hip_bf16.h>

// ---------------------------------------------------------------------------
// CDNA5 (gfx1250) bf16-WMMA implementation of MHA + RoPE forward.
// wave32; V_WMMA_F32_16X16X32_BF16; software-pipelined global->LDS staging.
// ---------------------------------------------------------------------------

typedef __attribute__((ext_vector_type(16))) __bf16 v16bf;
typedef __attribute__((ext_vector_type(8)))  float  v8f;

#define BB   2
#define LL   2048
#define DD   2048
#define HH   16
#define HD   128
#define QKVD 6144   // 3*DD

__device__ __forceinline__ v8f wmma_bf16(v16bf a, v16bf b, v8f c) {
  // (neg_a, A, neg_b, B, c_mod, C, reuse_a, reuse_b)
  return __builtin_amdgcn_wmma_f32_16x16x32_bf16(false, a, false, b, (short)0, c,
                                                 false, false);
}

__device__ __forceinline__ v8f zero8() {
  v8f z = {0.f, 0.f, 0.f, 0.f, 0.f, 0.f, 0.f, 0.f};
  return z;
}

// ===========================================================================
// Kernel 1: C[M,N] = A[M,K] * B[N,K]^T    (fp32 memory, bf16 WMMA math)
// Block: 256 threads = 8 waves. Block tile 128x128, K-chunk 32.
// Wave (wm,wn) owns a 32x64 C tile: 2 A-frags x 4 B-frags -> 8 WMMAs/chunk.
// Global loads for chunk k+1 are issued while WMMAs run on chunk k.
// ===========================================================================
__global__ __launch_bounds__(256) void gemm_xt_bf16(
    const float* __restrict__ A, const float* __restrict__ B,
    float* __restrict__ C, int M, int N, int K)
{
  __shared__ __bf16 sA[128][40];   // [m][k], +8 pad
  __shared__ __bf16 sB[128][40];   // [n][k], +8 pad

  const int tid  = threadIdx.x;
  const int lane = tid & 31;
  const int wave = tid >> 5;
  const int hi   = lane >> 4;      // 0: lanes 0-15, 1: lanes 16-31
  const int ln   = lane & 15;
  const int bm   = blockIdx.x * 128;
  const int bn   = blockIdx.y * 128;
  const int wm   = (wave & 3) * 32;    // wave's M offset inside block tile
  const int wn   = (wave >> 2) * 64;   // wave's N offset inside block tile

  // per-thread staging slots: 4 float4 per tile (128x32 fp32 / 256 threads)
  const int sr = (tid * 4) >> 5;       // constant across i via e = tid + i*256
  (void)sr;

  v8f acc[2][4];
#pragma unroll
  for (int mi = 0; mi < 2; ++mi)
#pragma unroll
    for (int nt = 0; nt < 4; ++nt) acc[mi][nt] = zero8();

  float4 ra[4], rb[4];
  // ---- initial global load (chunk 0): all loads in flight before any use
#pragma unroll
  for (int i = 0; i < 4; ++i) {
    int e = tid + i * 256;             // float4 index 0..1023
    int r = e >> 3;                    // row 0..127
    int c = (e & 7) << 2;              // col 0,4,...,28
    ra[i] = *(const float4*)&A[(size_t)(bm + r) * K + c];
    rb[i] = *(const float4*)&B[(size_t)(bn + r) * K + c];
  }

  for (int kk = 0; kk < K; kk += 32) {
    // ---- convert + store staged registers to LDS
#pragma unroll
    for (int i = 0; i < 4; ++i) {
      int e = tid + i * 256;
      int r = e >> 3;
      int c = (e & 7) << 2;
      sA[r][c + 0] = (__bf16)ra[i].x; sA[r][c + 1] = (__bf16)ra[i].y;
      sA[r][c + 2] = (__bf16)ra[i].z; sA[r][c + 3] = (__bf16)ra[i].w;
      sB[r][c + 0] = (__bf16)rb[i].x; sB[r][c + 1] = (__bf16)rb[i].y;
      sB[r][c + 2] = (__bf16)rb[i].z; sB[r][c + 3] = (__bf16)rb[i].w;
    }
    __syncthreads();

    // ---- prefetch next K-chunk into registers (overlaps with WMMA below)
    if (kk + 32 < K) {
#pragma unroll
      for (int i = 0; i < 4; ++i) {
        int e = tid + i * 256;
        int r = e >> 3;
        int c = (e & 7) << 2;
        ra[i] = *(const float4*)&A[(size_t)(bm + r) * K + kk + 32 + c];
        rb[i] = *(const float4*)&B[(size_t)(bn + r) * K + kk + 32 + c];
      }
    }

    // ---- preload all fragments, then back-to-back WMMAs
    v16bf af[2], bfr[4];
#pragma unroll
    for (int mi = 0; mi < 2; ++mi) {
      int m = wm + mi * 16 + ln;
#pragma unroll
      for (int e = 0; e < 16; ++e) {
        int k = (e < 8 ? e : e + 8) + hi * 8;   // ISA 16-bit A layout
        af[mi][e] = sA[m][k];
      }
    }
#pragma unroll
    for (int nt = 0; nt < 4; ++nt) {
      int n = wn + nt * 16 + ln;
#pragma unroll
      for (int e = 0; e < 16; ++e) {
        int k = (e < 8 ? e : e + 8) + hi * 8;   // ISA 16-bit B layout (same k order as A)
        bfr[nt][e] = sB[n][k];
      }
    }
#pragma unroll
    for (int nt = 0; nt < 4; ++nt) {
      acc[0][nt] = wmma_bf16(af[0], bfr[nt], acc[0][nt]);
      acc[1][nt] = wmma_bf16(af[1], bfr[nt], acc[1][nt]);
    }
    __syncthreads();
  }

  // ---- epilogue: C layout -> row = r + 8*hi, col = ln (per 16x16 subtile)
#pragma unroll
  for (int mi = 0; mi < 2; ++mi) {
#pragma unroll
    for (int nt = 0; nt < 4; ++nt) {
#pragma unroll
      for (int r = 0; r < 8; ++r) {
        int row = bm + wm + mi * 16 + r + hi * 8;
        int col = bn + wn + nt * 16 + ln;
        C[(size_t)row * N + col] = acc[mi][nt][r];
      }
    }
  }
}

// ===========================================================================
// Kernel 2: RoPE in place on the q and k thirds of the qkv buffer.
// One thread handles a (d, d+64) pair -> no in-place race.
// ===========================================================================
__global__ void rope_kernel(float* __restrict__ qkv,
                            const float* __restrict__ cosT,
                            const float* __restrict__ sinT)
{
  int idx = blockIdx.x * blockDim.x + threadIdx.x;
  int d  = idx & 63;   idx >>= 6;
  int h  = idx & 15;   idx >>= 4;
  int qk = idx & 1;    idx >>= 1;
  int l  = idx & 2047; idx >>= 11;
  int b  = idx;
  if (b >= BB) return;

  size_t base = ((size_t)(b * LL + l)) * QKVD + qk * DD + h * HD;
  float t0 = qkv[base + d];
  float t1 = qkv[base + d + 64];
  float c0 = cosT[l * HD + d],      s0 = sinT[l * HD + d];
  float c1 = cosT[l * HD + d + 64], s1 = sinT[l * HD + d + 64];
  qkv[base + d]      = t0 * c0 - t1 * s0;   // rotate_half: -t[d+64] for d<64
  qkv[base + d + 64] = t1 * c1 + t0 * s1;   // rotate_half:  t[d]    for d>=64
}

// ===========================================================================
// Kernel 3: flash attention. Block = (b, h, 128 q rows), 8 waves.
// Wave owns a 16-row q strip; loops over 32-key blocks up to the diagonal.
// Q in registers (bf16 A-frags); K row-major LDS; V transposed LDS so the
// B-fragments of both WMMAs are contiguous ds_load runs.
// ===========================================================================
__global__ __launch_bounds__(256) void flash_attn(
    const float* __restrict__ qkv, const unsigned char* __restrict__ keep,
    float* __restrict__ out)
{
  __shared__ __bf16 sK[32][136];        // [key][d]  (+8 pad)
  __shared__ __bf16 sVT[128][40];       // [d][key]  (+8 pad)
  __shared__ __bf16 sP[8][16][32];      // per-wave P tile (C->A relayout)
  __shared__ unsigned char sKeep[32];

  const int tid  = threadIdx.x;
  const int lane = tid & 31;
  const int wave = tid >> 5;
  const int hi   = lane >> 4;
  const int ln   = lane & 15;
  const int q0   = blockIdx.x * 128;
  const int h    = blockIdx.y;
  const int b    = blockIdx.z;
  const float scale = 0.08838834764831845f;  // 1/sqrt(128)

  // ---- preload Q as four bf16 A-fragments (head-dim chunks of 32)
  v16bf qf[4];
  {
    const int qm = q0 + wave * 16 + ln;                 // A-layout: m = ln
    size_t qbase = ((size_t)(b * LL + qm)) * QKVD + h * HD;
#pragma unroll
    for (int c = 0; c < 4; ++c) {
      const float* p = &qkv[qbase + c * 32 + hi * 8];
#pragma unroll
      for (int e = 0; e < 8; ++e) qf[c][e]     = (__bf16)p[e];       // k=k0..k0+7
#pragma unroll
      for (int e = 0; e < 8; ++e) qf[c][8 + e] = (__bf16)p[16 + e];  // k=k0+16..
    }
  }

  v8f acc[8];
#pragma unroll
  for (int i = 0; i < 8; ++i) acc[i] = zero8();
  float mrow[8], lrow[8];
#pragma unroll
  for (int r = 0; r < 8; ++r) { mrow[r] = -1e30f; lrow[r] = 0.f; }

  const int nkb = q0 / 32 + 4;          // causal: only blocks touching <= q0+127
  for (int kb = 0; kb < nkb; ++kb) {
    const int k0g = kb * 32;

    // ---- stage K (row-major) and V (transposed): load phase then store phase
    float4 kreg[4], vreg[4];
#pragma unroll
    for (int i = 0; i < 4; ++i) {
      int e = tid + i * 256;            // float4 index 0..1023 over 32x128
      int r = e >> 5;                   // key 0..31
      int c = (e & 31) << 2;            // dim 0,4,...,124
      size_t rowbase = ((size_t)(b * LL + k0g + r)) * QKVD + h * HD + c;
      kreg[i] = *(const float4*)&qkv[rowbase + DD];        // K third
      vreg[i] = *(const float4*)&qkv[rowbase + 2 * DD];    // V third
    }
#pragma unroll
    for (int i = 0; i < 4; ++i) {
      int e = tid + i * 256;
      int r = e >> 5;
      int c = (e & 31) << 2;
      sK[r][c + 0] = (__bf16)kreg[i].x; sK[r][c + 1] = (__bf16)kreg[i].y;
      sK[r][c + 2] = (__bf16)kreg[i].z; sK[r][c + 3] = (__bf16)kreg[i].w;
      sVT[c + 0][r] = (__bf16)vreg[i].x; sVT[c + 1][r] = (__bf16)vreg[i].y;
      sVT[c + 2][r] = (__bf16)vreg[i].z; sVT[c + 3][r] = (__bf16)vreg[i].w;
    }
    if (tid < 32) sKeep[tid] = keep[b * LL + k0g + tid];
    __syncthreads();

    // ---- S = Q K^T: preload 4 B-frags (2 d-chunks), then 4 WMMAs; x2
    v8f s0 = zero8(), s1 = zero8();
#pragma unroll
    for (int cc = 0; cc < 4; cc += 2) {
      v16bf bf0[2], bf1[2];
#pragma unroll
      for (int c2 = 0; c2 < 2; ++c2) {
#pragma unroll
        for (int e = 0; e < 16; ++e) {
          int d = (cc + c2) * 32 + (e < 8 ? e : e + 8) + hi * 8;  // B layout: table order, matches qf
          bf0[c2][e] = sK[ln][d];                // key = ln
          bf1[c2][e] = sK[16 + ln][d];           // key = 16 + ln
        }
      }
      s0 = wmma_bf16(qf[cc + 0], bf0[0], s0);
      s1 = wmma_bf16(qf[cc + 0], bf1[0], s1);
      s0 = wmma_bf16(qf[cc + 1], bf0[1], s0);
      s1 = wmma_bf16(qf[cc + 1], bf1[1], s1);
    }

    // ---- scale + causal + key-padding mask
    float sv[2][8];
    bool  ok[2][8];
#pragma unroll
    for (int nt = 0; nt < 2; ++nt) {
#pragma unroll
      for (int r = 0; r < 8; ++r) {
        int qrow = q0 + wave * 16 + r + hi * 8;
        int key  = k0g + nt * 16 + ln;
        float s  = (nt ? s1[r] : s0[r]) * scale;
        ok[nt][r] = (key <= qrow) && (sKeep[nt * 16 + ln] != 0);
        sv[nt][r] = ok[nt][r] ? s : -1e30f;
      }
    }

    // ---- online softmax (rows live in 16 lanes of one half; butterfly 1..8)
#pragma unroll
    for (int r = 0; r < 8; ++r) {
      float m = fmaxf(sv[0][r], sv[1][r]);
#pragma unroll
      for (int off = 1; off <= 8; off <<= 1)
        m = fmaxf(m, __shfl_xor(m, off, 32));
      float mn = fmaxf(mrow[r], m);
      float ef = __expf(mrow[r] - mn);
      float p0 = ok[0][r] ? __expf(sv[0][r] - mn) : 0.f;   // exact 0 for masked
      float p1 = ok[1][r] ? __expf(sv[1][r] - mn) : 0.f;
      float rs = p0 + p1;
#pragma unroll
      for (int off = 1; off <= 8; off <<= 1)
        rs += __shfl_xor(rs, off, 32);
      lrow[r] = lrow[r] * ef + rs;
      mrow[r] = mn;
#pragma unroll
      for (int dt = 0; dt < 8; ++dt) acc[dt][r] *= ef;
      sv[0][r] = p0; sv[1][r] = p1;
    }

    // ---- C-layout P -> LDS -> A-layout bf16 fragment (wave-private tile)
#pragma unroll
    for (int nt = 0; nt < 2; ++nt)
#pragma unroll
      for (int r = 0; r < 8; ++r)
        sP[wave][r + hi * 8][nt * 16 + ln] = (__bf16)sv[nt][r];

    v16bf pf;
#pragma unroll
    for (int e = 0; e < 16; ++e) {
      int k = (e < 8 ? e : e + 8) + hi * 8;
      pf[e] = sP[wave][ln][k];
    }

    // ---- O += P V: preload 4 V-frags, 4 WMMAs; x2  (8 out-dim subtiles)
#pragma unroll
    for (int dd = 0; dd < 8; dd += 4) {
      v16bf vf[4];
#pragma unroll
      for (int d2 = 0; d2 < 4; ++d2) {
#pragma unroll
        for (int e = 0; e < 16; ++e) {
          int k = (e < 8 ? e : e + 8) + hi * 8;  // key index within block
          vf[d2][e] = sVT[(dd + d2) * 16 + ln][k];
        }
      }
#pragma unroll
      for (int d2 = 0; d2 < 4; ++d2)
        acc[dd + d2] = wmma_bf16(pf, vf[d2], acc[dd + d2]);
    }
    __syncthreads();
  }

  // ---- epilogue: normalize, zero padded queries, write (B,L,H*HD)
#pragma unroll
  for (int r = 0; r < 8; ++r) {
    int qrow = q0 + wave * 16 + r + hi * 8;
    float inv = 1.f / lrow[r];
    float keepq = keep[b * LL + qrow] ? 1.f : 0.f;
#pragma unroll
    for (int dt = 0; dt < 8; ++dt) {
      int d = dt * 16 + ln;
      out[((size_t)(b * LL + qrow)) * DD + h * HD + d] = acc[dt][r] * inv * keepq;
    }
  }
}

// ===========================================================================
extern "C" void kernel_launch(void* const* d_in, const int* in_sizes, int n_in,
                              void* d_out, int out_size, void* d_ws, size_t ws_size,
                              hipStream_t stream)
{
  (void)in_sizes; (void)n_in; (void)out_size; (void)ws_size;

  const float*         x     = (const float*)d_in[0];
  const unsigned char* keep  = (const unsigned char*)d_in[1];  // jnp bool_ = 1 byte
  const float*         cosT  = (const float*)d_in[2];
  const float*         sinT  = (const float*)d_in[3];
  const float*         w_qkv = (const float*)d_in[4];
  const float*         w_out = (const float*)d_in[5];
  float*               outp  = (float*)d_out;

  float* qkv  = (float*)d_ws;                            // (B*L) x 6144  = 96 MB
  float* attn = qkv + (size_t)(BB * LL) * QKVD;          // (B*L) x 2048  = 32 MB

  // 1) QKV projection: (4096 x 6144) = x (4096 x 2048) * w_qkv^T
  {
    dim3 grid((BB * LL) / 128, QKVD / 128);
    gemm_xt_bf16<<<grid, 256, 0, stream>>>(x, w_qkv, qkv, BB * LL, QKVD, DD);
  }
  // 2) RoPE on q,k thirds (one thread per rotation pair)
  {
    int total = BB * LL * 2 * HH * 64;                   // 8,388,608
    rope_kernel<<<total / 256, 256, 0, stream>>>(qkv, cosT, sinT);
  }
  // 3) flash attention -> attn (B,L,D)
  {
    dim3 grid(LL / 128, HH, BB);
    flash_attn<<<grid, 256, 0, stream>>>(qkv, keep, attn);
  }
  // 4) output projection: out = attn * w_out^T
  {
    dim3 grid((BB * LL) / 128, DD / 128);
    gemm_xt_bf16<<<grid, 256, 0, stream>>>(attn, w_out, outp, BB * LL, DD, DD);
  }
}
